// Fp8DeepGEMMLinear_69879117906071
// MI455X (gfx1250) — 8-bit matrix operand order repaired and checked by reading, compiled, not run on hardware
//
#include <hip/hip_runtime.h>
#include <hip/hip_bf16.h>
#include <stdint.h>

#define GSZ 128
#define FP8MAX 448.0f
#define QEPS 1e-4f
#define MINSCALE (1e-4f / 448.0f)

typedef __attribute__((ext_vector_type(16))) int   v16i;
typedef __attribute__((ext_vector_type(8)))  float v8f;
typedef __attribute__((ext_vector_type(4)))  int   v4i;

__device__ __forceinline__ float bf2f(unsigned short u) {
    return __uint_as_float(((unsigned)u) << 16);
}

__device__ __forceinline__ unsigned sw_e4m3(float f) {
    unsigned u = __float_as_uint(f);
    unsigned s = (u >> 24) & 0x80u;
    unsigned e = (u >> 23) & 0xFFu;
    unsigned m = u & 0x7FFFFFu;
    int ex = (int)e - 127;
    if (e == 0xFFu) return s | 0x7Eu;
    if (e == 0u)    return s;
    if (ex >= 9)    return s | 0x7Eu;
    if (ex >= -6) {
        unsigned mm = m >> 20;
        unsigned rest = m & 0xFFFFFu;
        mm += (rest > 0x80000u) || (rest == 0x80000u && (mm & 1u));
        unsigned ee = (unsigned)(ex + 7);
        if (mm == 8u) { mm = 0u; ee += 1u; }
        if (ee >= 16u || (ee == 15u && mm == 7u)) return s | 0x7Eu;
        return s | (ee << 3) | mm;
    }
    if (ex < -10) return s;
    unsigned full = m | 0x800000u;
    int shift = 20 + (-6 - ex);
    unsigned mm = full >> shift;
    unsigned rest = full & ((1u << shift) - 1u);
    unsigned half = 1u << (shift - 1);
    mm += (rest > half) || (rest == half && (mm & 1u));
    if (mm >= 8u) return s | 0x08u;
    return s | mm;
}

__device__ __forceinline__ unsigned pack4_e4m3(float f0, float f1, float f2, float f3) {
    f0 = fminf(fmaxf(f0, -FP8MAX), FP8MAX);
    f1 = fminf(fmaxf(f1, -FP8MAX), FP8MAX);
    f2 = fminf(fmaxf(f2, -FP8MAX), FP8MAX);
    f3 = fminf(fmaxf(f3, -FP8MAX), FP8MAX);
#if defined(__has_builtin) && __has_builtin(__builtin_amdgcn_cvt_pk_fp8_f32)
    int w = __builtin_amdgcn_cvt_pk_fp8_f32(f0, f1, 0, false);
    w = __builtin_amdgcn_cvt_pk_fp8_f32(f2, f3, w, true);
    return (unsigned)w;
#else
    return sw_e4m3(f0) | (sw_e4m3(f1) << 8) | (sw_e4m3(f2) << 16) | (sw_e4m3(f3) << 24);
#endif
}

__device__ __forceinline__ void async_copy16(void* lds_dst, const void* gsrc) {
    unsigned loff = (unsigned)(unsigned long long)lds_dst;
    unsigned long long ga = (unsigned long long)gsrc;
    asm volatile("global_load_async_to_lds_b128 %0, %1, off"
                 :: "v"(loff), "v"(ga) : "memory");
}
__device__ __forceinline__ void wait_async0() {
    asm volatile("s_wait_asynccnt 0" ::: "memory");
}

__global__ void k_quant_x(const __hip_bfloat16* __restrict__ x,
                          unsigned char* __restrict__ xq,
                          float* __restrict__ xs_t,
                          int M, int K) {
    const int G = K / GSZ;
    long long wv = (long long)blockIdx.x * (blockDim.x >> 5) + (threadIdx.x >> 5);
    if (wv >= (long long)M * G) return;
    const int lane = threadIdx.x & 31;
    const int m = (int)(wv / G);
    const int g = (int)(wv % G);

    long long base = (long long)m * K + g * GSZ + lane * 4;
    unsigned long long v = *(const unsigned long long*)((const unsigned short*)x + base);
    float f0 = bf2f((unsigned short)(v));
    float f1 = bf2f((unsigned short)(v >> 16));
    float f2 = bf2f((unsigned short)(v >> 32));
    float f3 = bf2f((unsigned short)(v >> 48));

    float amax = fmaxf(fmaxf(fabsf(f0), fabsf(f1)), fmaxf(fabsf(f2), fabsf(f3)));
    #pragma unroll
    for (int off = 16; off > 0; off >>= 1)
        amax = fmaxf(amax, __shfl_xor(amax, off, 32));

    float s = fmaxf(fmaxf(amax, QEPS) / FP8MAX, MINSCALE);
    if (lane == 0) xs_t[(long long)g * M + m] = s;

    float inv = 1.0f / s;
    *(unsigned*)(xq + base) = pack4_e4m3(f0 * inv, f1 * inv, f2 * inv, f3 * inv);
}

__global__ void k_pack_w(const float* __restrict__ wq_f,
                         unsigned char* __restrict__ wq8,
                         long long total4) {
    long long i = (long long)blockIdx.x * blockDim.x + threadIdx.x;
    if (i >= total4) return;
    float4 f = ((const float4*)wq_f)[i];
    ((unsigned*)wq8)[i] = pack4_e4m3(f.x, f.y, f.z, f.w);
}

__global__ __launch_bounds__(256)
void k_gemm(const unsigned char* __restrict__ xq,
            const unsigned char* __restrict__ wq8,
            const float* __restrict__ xs_t,
            const float* __restrict__ w_scales,
            const float* __restrict__ bias,
            __hip_bfloat16* __restrict__ out,
            int M, int N, int K) {
    const int NKB = K / GSZ;
    const int GB  = K / GSZ;
    const int nBase = blockIdx.x * 128;
    const int mBase = blockIdx.y * 128;
    const int nb = nBase / GSZ;

    __shared__ __align__(16) unsigned char As[2][128 * 128];
    __shared__ __align__(16) unsigned char Bs[2][128 * 128];

    const int t     = threadIdx.x;
    const int lane  = t & 31;
    const int wv    = t >> 5;
    const int waveM = wv >> 1;
    const int waveN = wv & 1;
    const int h     = lane >> 4;
    const int l16   = lane & 15;

    const int rr = t >> 3;
    const int cc = (t & 7) << 4;

    const unsigned char* xA = xq  + (long long)mBase * K;
    const unsigned char* wB = wq8 + (long long)nBase * K;

    v8f acc[2][4];
    #pragma unroll
    for (int a = 0; a < 2; ++a)
        #pragma unroll
        for (int b = 0; b < 4; ++b)
            acc[a][b] = (v8f)(0.0f);
    const v8f czero = (v8f)(0.0f);

    #pragma unroll
    for (int i = 0; i < 4; ++i) {
        int r = rr + i * 32;
        async_copy16(&As[0][r * 128 + cc], xA + (long long)r * K + cc);
        async_copy16(&Bs[0][r * 128 + cc], wB + (long long)r * K + cc);
    }

    for (int kb = 0; kb < NKB; ++kb) {
        const int cur = kb & 1;
        wait_async0();
        __syncthreads();

        if (kb + 1 < NKB) {
            const int nxt = cur ^ 1;
            long long kOff = (long long)(kb + 1) * GSZ;
            #pragma unroll
            for (int i = 0; i < 4; ++i) {
                int r = rr + i * 32;
                async_copy16(&As[nxt][r * 128 + cc], xA + (long long)r * K + kOff + cc);
                async_copy16(&Bs[nxt][r * 128 + cc], wB + (long long)r * K + kOff + cc);
            }
        }

        const float wsg = w_scales[nb * GB + kb];

        v8f sv[2];
        #pragma unroll
        for (int tm = 0; tm < 2; ++tm) {
            int m0 = mBase + waveM * 32 + tm * 16 + 8 * h;
            const float4* sp = (const float4*)(xs_t + (long long)kb * M + m0);
            float4 s0 = sp[0], s1 = sp[1];
            sv[tm][0] = s0.x * wsg; sv[tm][1] = s0.y * wsg;
            sv[tm][2] = s0.z * wsg; sv[tm][3] = s0.w * wsg;
            sv[tm][4] = s1.x * wsg; sv[tm][5] = s1.y * wsg;
            sv[tm][6] = s1.z * wsg; sv[tm][7] = s1.w * wsg;
        }

        v16i afr[2];
        #pragma unroll
        for (int tm = 0; tm < 2; ++tm) {
            const unsigned char* ap =
                &As[cur][(waveM * 32 + tm * 16 + l16) * 128 + h * 16];
            unsigned long long* a64 = (unsigned long long*)&afr[tm];
            #pragma unroll
            for (int c = 0; c < 8; ++c) {
                int koff = ((c >> 1) << 5) + ((c & 1) << 3);
                a64[c] = *(const unsigned long long*)(ap + koff);
            }
        }

        #pragma unroll
        for (int tn = 0; tn < 4; ++tn) {
            const unsigned char* bp =
                &Bs[cur][(waveN * 64 + tn * 16 + l16) * 128 + h * 16];
            v16i bfr;
            v4i* b4 = (v4i*)&bfr;
            #pragma unroll
            for (int gq = 0; gq < 4; ++gq)
                b4[gq] = *(const v4i*)(bp + (gq << 5));

            #pragma unroll
            for (int tm = 0; tm < 2; ++tm) {
                v8f part = __builtin_amdgcn_wmma_f32_16x16x128_fp8_fp8(
                    afr[tm], bfr, (short)0, czero, false, false);
                acc[tm][tn] += sv[tm] * part;
            }
        }
    }

    #pragma unroll
    for (int tm = 0; tm < 2; ++tm) {
        int m0 = mBase + waveM * 32 + tm * 16 + 8 * h;
        #pragma unroll
        for (int tn = 0; tn < 4; ++tn) {
            int n = nBase + waveN * 64 + tn * 16 + l16;
            float bv = bias[n];
            #pragma unroll
            for (int r = 0; r < 8; ++r) {
                out[(long long)(m0 + r) * N + n] =
                    __float2bfloat16(acc[tm][tn][r] + bv);
            }
        }
    }
}

extern "C" void kernel_launch(void* const* d_in, const int* in_sizes, int n_in,
                              void* d_out, int out_size, void* d_ws, size_t ws_size,
                              hipStream_t stream) {
    const __hip_bfloat16* x  = (const __hip_bfloat16*)d_in[0];
    const float* w_q         = (const float*)d_in[1];
    const float* w_scales    = (const float*)d_in[2];
    const float* bias        = (const float*)d_in[3];
    __hip_bfloat16* out      = (__hip_bfloat16*)d_out;

    const int N = in_sizes[3];
    const int K = in_sizes[1] / N;
    const int M = in_sizes[0] / K;
    const int G = K / GSZ;

    unsigned char* ws  = (unsigned char*)d_ws;
    unsigned char* xq  = ws;
    unsigned char* wq8 = ws + (size_t)M * K;
    float* xs_t = (float*)(ws + (size_t)M * K + (size_t)N * K);

    long long groups = (long long)M * G;
    k_quant_x<<<dim3((unsigned)((groups + 7) / 8)), dim3(256), 0, stream>>>(
        x, xq, xs_t, M, K);

    long long w4 = (long long)N * K / 4;
    k_pack_w<<<dim3((unsigned)((w4 + 255) / 256)), dim3(256), 0, stream>>>(
        w_q, wq8, w4);

    k_gemm<<<dim3(N / 128, M / 128), dim3(256), 0, stream>>>(
        xq, wq8, xs_t, w_scales, bias, out, M, N, K);
}
